// CrossAttention_85358180040962
// MI455X (gfx1250) — matrix operand order repaired and checked by reading, not compiled, not run on hardware
//
#include <hip/hip_runtime.h>
#include <hip/hip_bf16.h>

// ---- CDNA5 (gfx1250) wave32 WMMA types ----
typedef __attribute__((ext_vector_type(16))) __bf16 v16bf;
typedef __attribute__((ext_vector_type(8)))  __bf16 v8bf;
typedef __attribute__((ext_vector_type(8)))  float  v8f;

#define DIM   4096
#define HD    128
#define NH    32
#define KV    1024
#define NTOK  3072

// Assemble a WMMA 16-element bf16 fragment from two 16-byte chunks.
__device__ __forceinline__ v16bf ld_frag(const __bf16* __restrict__ p0,
                                         const __bf16* __restrict__ p1) {
    union { v16bf v; v8bf h[2]; } u;
    u.h[0] = *(const v8bf*)p0;
    u.h[1] = *(const v8bf*)p1;
    return u.v;
}

// =====================================================================
// f32 -> bf16 stream convert (8 elems/thread, b128 in / b128 out)
// =====================================================================
__global__ __launch_bounds__(256)
void cvt_f32_bf16(const float* __restrict__ in, __bf16* __restrict__ out, int n8)
{
    const int i = blockIdx.x * 256 + threadIdx.x;
    if (i >= n8) return;
    union { float4 v; float f[4]; } a, b;
    a.v = *(const float4*)(in + (size_t)i * 8);
    b.v = *(const float4*)(in + (size_t)i * 8 + 4);
    v8bf o;
#pragma unroll
    for (int e = 0; e < 4; ++e) { o[e] = (__bf16)a.f[e]; o[4 + e] = (__bf16)b.f[e]; }
    *(v8bf*)(out + (size_t)i * 8) = o;
}

// =====================================================================
// V transpose+convert: [bh][kv][d] f32  ->  [bh][d][kv] bf16
// 32x32 tiles through LDS.
// =====================================================================
__global__ __launch_bounds__(256)
void transpose_v(const float* __restrict__ V, __bf16* __restrict__ Vt)
{
    __shared__ float tile[32][33];
    const int kv0 = blockIdx.x * 32;
    const int d0  = blockIdx.y * 32;
    const int bh  = blockIdx.z;
    const int tx = threadIdx.x & 31;      // inner (coalesced) index
    const int ty = threadIdx.x >> 5;      // 8 rows per pass
    const float* Vb = V + (size_t)bh * KV * HD;
    __bf16* Tb = Vt + (size_t)bh * HD * KV;
#pragma unroll
    for (int p = 0; p < 4; ++p) {
        const int kr = ty + p * 8;
        tile[kr][tx] = Vb[(size_t)(kv0 + kr) * HD + d0 + tx];
    }
    __syncthreads();
#pragma unroll
    for (int p = 0; p < 4; ++p) {
        const int dr = ty + p * 8;
        Tb[(size_t)(d0 + dr) * KV + kv0 + tx] = (__bf16)tile[tx][dr];
    }
}

// =====================================================================
// GEMM: C[M,N] = A[M,K] * B[N,K]^T   (A,B bf16 row-major, K-contiguous)
// 128x128 block tile, 4 waves, 64x64/wave, bf16 WMMA / f32 accum.
// EPI==0: plain f32 store.  EPI==1: per-row RMSNorm (head_dim = block's
// 128 cols) * qnw, stored bf16.
// =====================================================================
template<int EPI>
__global__ __launch_bounds__(128)
void gemm_nt_bf16(const __bf16* __restrict__ A, const __bf16* __restrict__ B,
                  void* __restrict__ Cout, const float* __restrict__ qnw,
                  int M, int N, int K)
{
    __shared__ float rowss[128][2];
    const int lane  = threadIdx.x & 31;
    const int wave  = threadIdx.x >> 5;
    const int wm = wave >> 1, wn = wave & 1;
    const int mBase = blockIdx.x * 128 + wm * 64;
    const int nBase = blockIdx.y * 128 + wn * 64;
    const int lr = lane & 15;       // row/col within 16-tile
    const int hi = lane >> 4;       // lane-half -> K sub-range

    v8f acc[4][4];
#pragma unroll
    for (int i = 0; i < 4; ++i)
#pragma unroll
        for (int j = 0; j < 4; ++j) acc[i][j] = {};

    for (int k0 = 0; k0 < K; k0 += 32) {
        v16bf a[4], b[4];
#pragma unroll
        for (int t = 0; t < 4; ++t) {
            const __bf16* ap = A + (size_t)(mBase + t * 16 + lr) * K + k0 + hi * 8;
            a[t] = ld_frag(ap, ap + 16);
            const __bf16* bp = B + (size_t)(nBase + t * 16 + lr) * K + k0 + hi * 8;
            b[t] = ld_frag(bp, bp + 16);
        }
#pragma unroll
        for (int i = 0; i < 4; ++i)
#pragma unroll
            for (int j = 0; j < 4; ++j)
                acc[i][j] = __builtin_amdgcn_wmma_f32_16x16x32_bf16(
                    false, a[i], false, b[j], (short)0, acc[i][j], false, false);
    }

    if (EPI == 0) {
        float* C = (float*)Cout;
#pragma unroll
        for (int i = 0; i < 4; ++i)
#pragma unroll
            for (int j = 0; j < 4; ++j)
#pragma unroll
                for (int r = 0; r < 8; ++r) {
                    const int row = mBase + i * 16 + r + hi * 8;
                    const int col = nBase + j * 16 + lr;
                    C[(size_t)row * N + col] = acc[i][j][r];
                }
    } else {
        // ---- fused RMSNorm over the block's 128 cols (one head) ----
#pragma unroll
        for (int i = 0; i < 4; ++i)
#pragma unroll
            for (int r = 0; r < 8; ++r) {
                float s = 0.f;
#pragma unroll
                for (int j = 0; j < 4; ++j) { float v = acc[i][j][r]; s += v * v; }
#pragma unroll
                for (int o = 1; o < 16; o <<= 1) s += __shfl_xor(s, o, 32);
                if (lr == 0) {
                    const int lrow = wm * 64 + i * 16 + r + hi * 8;
                    rowss[lrow][wn] = s;
                }
            }
        __syncthreads();
        __bf16* C = (__bf16*)Cout;
#pragma unroll
        for (int i = 0; i < 4; ++i)
#pragma unroll
            for (int r = 0; r < 8; ++r) {
                const int lrow = wm * 64 + i * 16 + r + hi * 8;
                const float tot = rowss[lrow][0] + rowss[lrow][1];
                const float rs = rsqrtf(tot * (1.0f / 128.0f) + 1e-5f);
#pragma unroll
                for (int j = 0; j < 4; ++j) {
                    const int lcol = wn * 64 + j * 16 + lr;      // head-dim index
                    const int row = blockIdx.x * 128 + lrow;
                    const int col = blockIdx.y * 128 + lcol;
                    C[(size_t)row * N + col] = (__bf16)(acc[i][j][r] * rs * qnw[lcol]);
                }
            }
    }
}

// =====================================================================
// Attention over packed tokens: one wave per (16-token tile, head).
// All operands bf16 (qn, K, V^T). S -> LDS f32, softmax, O = P*V.
// =====================================================================
__global__ __launch_bounds__(32)
void xattn(const __bf16* __restrict__ qn, const __bf16* __restrict__ kb,
           const __bf16* __restrict__ vt, const float* __restrict__ mask,
           const float* __restrict__ rowmask, const int* __restrict__ seq_lens,
           __bf16* __restrict__ attn_out)
{
    __shared__ float S[16 * KV];          // 64 KB score/prob buffer
    const int t = blockIdx.x;             // token tile (never crosses batch)
    const int h = blockIdx.y;
    const int lane = threadIdx.x;
    const int lr = lane & 15, hi = lane >> 4;
    const int tok0 = t * 16;

    const int l0 = seq_lens[0], l1 = seq_lens[1], l2 = seq_lens[2];
    int b = 0, start = 0;
    if (tok0 >= l0)           { b = 1; start = l0; }
    if (tok0 >= l0 + l1)      { b = 2; start = l0 + l1; }
    if (tok0 >= l0 + l1 + l2) { b = 3; start = l0 + l1 + l2; }
    const int p0 = tok0 - start;

    const float scale = 0.08838834764831845f;   // 1/sqrt(128)

    // Q fragments, resident for the whole kernel
    v16bf aq[4];
#pragma unroll
    for (int j = 0; j < 4; ++j) {
        const __bf16* qp = qn + (size_t)(tok0 + lr) * DIM + h * HD + j * 32 + hi * 8;
        aq[j] = ld_frag(qp, qp + 16);
    }

    const __bf16* Kb = kb + ((size_t)b * NH + h) * KV * HD;   // [kv][d]
    const __bf16* Vt = vt + ((size_t)b * NH + h) * HD * KV;   // [d][kv]
    const float*  Mb = mask + ((size_t)b * 1024 + p0) * KV;

    // ---- Phase 1: S = Q K^T * scale + mask ----
    for (int kt = 0; kt < KV / 16; ++kt) {
        v8f s = {};
#pragma unroll
        for (int j = 0; j < 4; ++j) {
            const __bf16* kp = Kb + (size_t)(kt * 16 + lr) * HD + j * 32 + hi * 8;
            s = __builtin_amdgcn_wmma_f32_16x16x32_bf16(
                    false, aq[j], false, ld_frag(kp, kp + 16), (short)0, s, false, false);
        }
#pragma unroll
        for (int r = 0; r < 8; ++r) {
            const int row = r + hi * 8;
            const int col = kt * 16 + lr;
            S[row * KV + col] = s[r] * scale + Mb[(size_t)row * KV + col];
        }
    }
    __syncthreads();

    // ---- Phase 2: softmax (row lr; halves split across lane pairs) ----
    {
        float* Srow = S + lr * KV + hi * (KV / 2);
        float m = -3.0e38f;
        for (int c = 0; c < KV / 2; ++c) m = fmaxf(m, Srow[c]);
        m = fmaxf(m, __shfl_xor(m, 16, 32));
        float sum = 0.f;
        for (int c = 0; c < KV / 2; ++c) { float e = __expf(Srow[c] - m); Srow[c] = e; sum += e; }
        sum += __shfl_xor(sum, 16, 32);
        const float inv = 1.0f / sum;
        for (int c = 0; c < KV / 2; ++c) Srow[c] *= inv;
    }
    __syncthreads();

    // ---- Phase 3: O = P * V ----
    v8f o[8];
#pragma unroll
    for (int j = 0; j < 8; ++j) o[j] = {};
    for (int k0 = 0; k0 < KV; k0 += 32) {
        // P fragment from LDS (f32 -> bf16; only remaining conversion)
        const float* pp = S + lr * KV + k0 + hi * 8;
        v16bf pa;
#pragma unroll
        for (int e = 0; e < 8; ++e) { pa[e] = (__bf16)pp[e]; pa[8 + e] = (__bf16)pp[16 + e]; }
#pragma unroll
        for (int j = 0; j < 8; ++j) {
            // B fragment: col n = j*16+lr of V; contraction contiguous in Vt
            const __bf16* vp = Vt + (size_t)(j * 16 + lr) * KV + k0 + hi * 8;
            o[j] = __builtin_amdgcn_wmma_f32_16x16x32_bf16(
                       false, pa, false, ld_frag(vp, vp + 16), (short)0, o[j], false, false);
        }
    }

    // ---- Epilogue: row mask, store bf16 packed ----
#pragma unroll
    for (int j = 0; j < 8; ++j)
#pragma unroll
        for (int r = 0; r < 8; ++r) {
            const int row = r + hi * 8;
            const int col = j * 16 + lr;
            const float rm = rowmask[tok0 + row];
            attn_out[(size_t)(tok0 + row) * DIM + h * HD + col] = (__bf16)(o[j][r] * rm);
        }
}

// =====================================================================
extern "C" void kernel_launch(void* const* d_in, const int* in_sizes, int n_in,
                              void* d_out, int out_size, void* d_ws, size_t ws_size,
                              hipStream_t stream) {
    (void)in_sizes; (void)n_in; (void)out_size; (void)ws_size;
    const float* x        = (const float*)d_in[0];
    const float* amask    = (const float*)d_in[1];
    const float* rowmask  = (const float*)d_in[2];
    const float* cache    = (const float*)d_in[3];
    const int*   seq_lens = (const int*)d_in[5];
    const float* wq       = (const float*)d_in[6];
    const float* wo       = (const float*)d_in[7];
    const float* qnw      = (const float*)d_in[8];
    float*       out      = (float*)d_out;

    char* ws = (char*)d_ws;
    const size_t NE = (size_t)NTOK * DIM;            // 12,582,912
    const size_t WE = (size_t)DIM * DIM;             // 16,777,216
    size_t off = 0;
    __bf16* qn   = (__bf16*)(ws + off); off += NE * 2;   // 24 MB
    __bf16* ao   = (__bf16*)(ws + off); off += NE * 2;   // 24 MB
    __bf16* xbf  = (__bf16*)(ws + off); off += NE * 2;   // 24 MB
    __bf16* wqbf = (__bf16*)(ws + off); off += WE * 2;   // 32 MB
    __bf16* wobf = (__bf16*)(ws + off); off += WE * 2;   // 32 MB
    __bf16* kbf  = (__bf16*)(ws + off); off += WE * 2;   // 32 MB
    __bf16* vtbf = (__bf16*)(ws + off); off += WE * 2;   // 32 MB

    const float* kc = cache;
    const float* vc = cache + WE;                    // [2,B,H,kv,d] -> V half

    // 1) bf16 staging (one-pass streams)
    cvt_f32_bf16<<<dim3((NE / 8 + 255) / 256), 256, 0, stream>>>(x, xbf, (int)(NE / 8));
    cvt_f32_bf16<<<dim3((WE / 8 + 255) / 256), 256, 0, stream>>>(wq, wqbf, (int)(WE / 8));
    cvt_f32_bf16<<<dim3((WE / 8 + 255) / 256), 256, 0, stream>>>(wo, wobf, (int)(WE / 8));
    cvt_f32_bf16<<<dim3((WE / 8 + 255) / 256), 256, 0, stream>>>(kc, kbf, (int)(WE / 8));
    transpose_v<<<dim3(KV / 32, HD / 32, 4 * NH), 256, 0, stream>>>(vc, vtbf);

    // 2) Q-proj GEMM with fused RMSNorm -> bf16 qn
    dim3 gG(NTOK / 128, DIM / 128);                  // 24 x 32
    gemm_nt_bf16<1><<<gG, 128, 0, stream>>>(xbf, wqbf, qn, qnw, NTOK, DIM, DIM);

    // 3) attention on packed tokens -> bf16 ao (row mask applied)
    xattn<<<dim3(NTOK / 16, NH), 32, 0, stream>>>(qn, kbf, vtbf, amask, rowmask,
                                                  seq_lens, ao);

    // 4) output projection -> f32 d_out
    gemm_nt_bf16<0><<<gG, 128, 0, stream>>>(ao, wobf, out, nullptr, NTOK, DIM, DIM);
}
